// VanillaRNN_1460288880744
// MI455X (gfx1250) — matrix operand order repaired and checked by reading, not compiled, not run on hardware
//
#include <hip/hip_runtime.h>
#include <math.h>

// ---------------------------------------------------------------------------
// Vanilla RNN scan fused kernel for MI455X (gfx1250, wave32, WMMA).
//
//   h_t = tanh(x_t @ W_xh + h_{t-1} @ W_hh + b_h)   (1024 sequential steps)
//   y   = h_final @ W_hy + b_y
//
// bf16 WMMA (v_wmma_f32_16x16x32_bf16), f32 accumulate. Weight re-fetch from
// L2 is the binding constraint -> 32 batch rows (2 M-tiles) per WG so every
// B-fragment fetched feeds 2 WMMAs (32 FLOP/byte). 8 WGs x 32 waves; wave w
// owns hidden columns [16w, 16w+16). h lives in LDS (bf16, single buffer,
// two barriers/step). Weight indices carry an opaque zero offset (SGPR asm
// barrier, redefined per step): loads stay inside the time loop (no
// hoist+spill to scratch), keep kernarg provenance (-> global_load_b128,
// LOADcnt only, no flat/DScnt coupling), and hit the L2-resident (<1 MB)
// packed weights.
// ---------------------------------------------------------------------------

#define BATCH   256
#define SEQ     1024
#define IN_DIM  128
#define HIDDEN  512
#define CLASSES 128

#define BROWS   32              // batch rows per WG (two WMMA M-tiles)
#define HPAD    8
#define XPAD    8

typedef __attribute__((ext_vector_type(16))) __bf16 v16bf;
typedef __attribute__((ext_vector_type(8)))  float  v8f;

// --- Pack a KxN row-major f32 matrix into WMMA B-fragment-ordered bf16 tiles.
// Tile (kt,nt) = 32 lanes x 16 bf16. Lane l holds column n = nt*16 + (l&15),
// rows k = kt*32 + (i>>3)*16 + (l>>4)*8 + (i&7), i=0..15  (matches 32x16 bf16 B layout).
// Linear tile index = nt*KT + kt (nt-major so a wave's K-walk is contiguous).
__global__ void pack_b_frags(const float* __restrict__ W, __bf16* __restrict__ out,
                             int K, int N) {
  int idx = blockIdx.x * blockDim.x + threadIdx.x;
  if (idx >= K * N) return;
  int i    = idx & 15;
  int lane = (idx >> 4) & 31;
  int tile = idx >> 9;
  int KT   = K >> 5;
  int kt   = tile % KT;
  int nt   = tile / KT;
  int n = nt * 16 + (lane & 15);
  int k = kt * 32 + (i >> 3) * 16 + (lane >> 4) * 8 + (i & 7);
  out[idx] = (__bf16)W[k * N + n];
}

// Load a 16x32 bf16 A-fragment from a row of shared memory.
// p points at row m (= lane&15 within the M-tile), element kt*32 + (lane>>4)*8.
// Chunks: k0..k0+7 and k0+16..k0+23  => two 16-byte LDS loads.
__device__ __forceinline__ v16bf load_a_frag(const __bf16* p) {
  union { uint4 q[2]; v16bf v; } u;
  u.q[0] = *(const uint4*)p;
  u.q[1] = *(const uint4*)(p + 16);
  return u.v;
}

__global__ __launch_bounds__(1024)
void rnn_scan_kernel(const float* __restrict__ x,
                     const __bf16* __restrict__ Wxh_pk,
                     const __bf16* __restrict__ Whh_pk,
                     const __bf16* __restrict__ Why_pk,
                     const float*  __restrict__ b_h,
                     const float*  __restrict__ b_y,
                     float*        __restrict__ out) {
  __shared__ __bf16 hbuf[BROWS][HIDDEN + HPAD];      // 33 KB, single buffer
  __shared__ __bf16 xbuf[2][BROWS][IN_DIM + XPAD];   // 17 KB, double buffered

  const int tid    = threadIdx.x;
  const int lane   = tid & 31;
  const int wave   = tid >> 5;          // 0..31, owns hidden cols [16w,16w+16)
  const int batch0 = blockIdx.x * BROWS;

  const int col  = lane & 15;           // N index in C/D, M row in A
  const int half = lane >> 4;           // K-chunk / M-half select

  const float bh = b_h[wave * 16 + col];

  // h_0 = 0 ; stage x_0
  for (int e = tid; e < BROWS * (HIDDEN + HPAD); e += blockDim.x)
    ((__bf16*)hbuf)[e] = (__bf16)0.f;
  for (int e = tid; e < BROWS * IN_DIM; e += blockDim.x) {
    int r = e >> 7, c = e & (IN_DIM - 1);
    xbuf[0][r][c] = (__bf16)x[((size_t)(batch0 + r) * SEQ) * IN_DIM + c];
  }
  __syncthreads();

  int cur = 0;
  for (int t = 0; t < SEQ; ++t) {
    const int nxt = cur ^ 1;

    // Stage x_{t+1} (f32 -> bf16) while computing step t.
    if (t + 1 < SEQ) {
      for (int e = tid; e < BROWS * IN_DIM; e += blockDim.x) {
        int r = e >> 7, c = e & (IN_DIM - 1);
        xbuf[nxt][r][c] =
            (__bf16)x[((size_t)(batch0 + r) * SEQ + (t + 1)) * IN_DIM + c];
      }
    }

    // Opaque zero offset, redefined every iteration in an SGPR: weight-load
    // addresses are not loop-invariant (no hoist + scratch spill), remain
    // uniform-based, and keep kernarg provenance so they lower to
    // global_load_b128 (LOADcnt only) hitting L2.
    size_t zx = 0, zh = 0;
    asm volatile("" : "+s"(zx), "+s"(zh));
    const __bf16* wx = Wxh_pk + zx;
    const __bf16* wh = Whh_pk + zh;

    v8f acc0 = {};   // M-tile 0: batch rows 0..15
    v8f acc1 = {};   // M-tile 1: batch rows 16..31

    // xh contribution: K = IN_DIM (4 kt x 2 M-tiles = 8 WMMAs)
    #pragma unroll
    for (int kt = 0; kt < IN_DIM / 32; ++kt) {
      v16bf b  = *(const v16bf*)(wx +
                 (((size_t)wave * (IN_DIM / 32) + kt) * 32 + lane) * 16);
      v16bf a0 = load_a_frag(&xbuf[cur][col     ][kt * 32 + half * 8]);
      v16bf a1 = load_a_frag(&xbuf[cur][col + 16][kt * 32 + half * 8]);
      acc0 = __builtin_amdgcn_wmma_f32_16x16x32_bf16(false, a0, false, b,
                                                     (short)0, acc0, false, false);
      acc1 = __builtin_amdgcn_wmma_f32_16x16x32_bf16(false, a1, false, b,
                                                     (short)0, acc1, false, false);
    }
    // h @ W_hh contribution: K = HIDDEN (16 kt x 2 M-tiles = 32 WMMAs)
    #pragma unroll
    for (int kt = 0; kt < HIDDEN / 32; ++kt) {
      v16bf b  = *(const v16bf*)(wh +
                 (((size_t)wave * (HIDDEN / 32) + kt) * 32 + lane) * 16);
      v16bf a0 = load_a_frag(&hbuf[col     ][kt * 32 + half * 8]);
      v16bf a1 = load_a_frag(&hbuf[col + 16][kt * 32 + half * 8]);
      acc0 = __builtin_amdgcn_wmma_f32_16x16x32_bf16(false, a0, false, b,
                                                     (short)0, acc0, false, false);
      acc1 = __builtin_amdgcn_wmma_f32_16x16x32_bf16(false, a1, false, b,
                                                     (short)0, acc1, false, false);
    }

    // All reads of hbuf are done; wait for every wave before overwriting it.
    __syncthreads();

    // bias + tanh, write h_{t+1} (C/D layout: vgpr r -> M = r + 8*half)
    #pragma unroll
    for (int r = 0; r < 8; ++r) {
      int m = r + half * 8;
      hbuf[m     ][wave * 16 + col] = (__bf16)tanhf(acc0[r] + bh);
      hbuf[m + 16][wave * 16 + col] = (__bf16)tanhf(acc1[r] + bh);
    }
    __syncthreads();
    cur = nxt;
  }

  // y = h_final @ W_hy + b_y : CLASSES/16 = 8 N-tiles, waves 0..7
  if (wave < CLASSES / 16) {
    const float by = b_y[wave * 16 + col];
    v8f acc0 = {};
    v8f acc1 = {};
    #pragma unroll
    for (int kt = 0; kt < HIDDEN / 32; ++kt) {
      v16bf b  = *(const v16bf*)(Why_pk +
                 (((size_t)wave * (HIDDEN / 32) + kt) * 32 + lane) * 16);
      v16bf a0 = load_a_frag(&hbuf[col     ][kt * 32 + half * 8]);
      v16bf a1 = load_a_frag(&hbuf[col + 16][kt * 32 + half * 8]);
      acc0 = __builtin_amdgcn_wmma_f32_16x16x32_bf16(false, a0, false, b,
                                                     (short)0, acc0, false, false);
      acc1 = __builtin_amdgcn_wmma_f32_16x16x32_bf16(false, a1, false, b,
                                                     (short)0, acc1, false, false);
    }
    #pragma unroll
    for (int r = 0; r < 8; ++r) {
      int m = r + half * 8;
      out[(size_t)(batch0 + m     ) * CLASSES + wave * 16 + col] = acc0[r] + by;
      out[(size_t)(batch0 + m + 16) * CLASSES + wave * 16 + col] = acc1[r] + by;
    }
  }
}

extern "C" void kernel_launch(void* const* d_in, const int* in_sizes, int n_in,
                              void* d_out, int out_size, void* d_ws, size_t ws_size,
                              hipStream_t stream) {
  // setup_inputs() dict order: x, W_hh, W_xh, W_hy, b_h, b_y (all f32)
  const float* x    = (const float*)d_in[0];
  const float* W_hh = (const float*)d_in[1];
  const float* W_xh = (const float*)d_in[2];
  const float* W_hy = (const float*)d_in[3];
  const float* b_h  = (const float*)d_in[4];
  const float* b_y  = (const float*)d_in[5];

  // Workspace: pre-packed bf16 B-fragment weights (< 1 MB total, L2-resident).
  __bf16* Wxh_pk = (__bf16*)d_ws;                                   // 128x512
  __bf16* Whh_pk = (__bf16*)((char*)d_ws + (size_t)IN_DIM * HIDDEN * 2);
  __bf16* Why_pk = (__bf16*)((char*)d_ws + (size_t)IN_DIM * HIDDEN * 2
                                         + (size_t)HIDDEN * HIDDEN * 2);

  pack_b_frags<<<(IN_DIM  * HIDDEN  + 255) / 256, 256, 0, stream>>>(W_xh, Wxh_pk, IN_DIM, HIDDEN);
  pack_b_frags<<<(HIDDEN  * HIDDEN  + 255) / 256, 256, 0, stream>>>(W_hh, Whh_pk, HIDDEN, HIDDEN);
  pack_b_frags<<<(HIDDEN  * CLASSES + 255) / 256, 256, 0, stream>>>(W_hy, Why_pk, HIDDEN, CLASSES);

  rnn_scan_kernel<<<BATCH / BROWS, 1024, 0, stream>>>(
      x, Wxh_pk, Whh_pk, Why_pk, b_h, b_y, (float*)d_out);
}
